// LSTM_44263932953053
// MI455X (gfx1250) — matrix operand order repaired and checked by reading, not compiled, not run on hardware
//
#include <hip/hip_runtime.h>
#include <hip/hip_bf16.h>
#include <stdint.h>

// ---------------------------------------------------------------------------
// LSTM for MI455X (gfx1250): fused input-projection + recurrence, bf16 WMMA.
//   B=32, T=1024, I=512, H=1024; gates = W @ [x_t ; h_t] + b  (K = 1536)
// Persistent kernel: 64 WGs x 256 threads, one grid barrier per timestep.
// Weights pre-packed bf16 (12 MB, L2-resident); h ping-pong bf16 in global.
// ---------------------------------------------------------------------------

typedef __attribute__((ext_vector_type(16))) __bf16 v16bf;
typedef __attribute__((ext_vector_type(8)))  float  v8f;

#define BATCH   32
#define SEQ     1024
#define IDIM    512
#define HDIM    1024
#define KDIM    (IDIM + HDIM)   // 1536
#define GDIM    (4 * HDIM)      // 4096
#define NBLOCKS 64

__device__ __forceinline__ unsigned short f32_to_bf16(float f) {
    union { float f; unsigned u; } v; v.f = f;
    unsigned u = v.u;
    unsigned r = u + 0x7FFFu + ((u >> 16) & 1u);   // round-to-nearest-even
    return (unsigned short)(r >> 16);
}

// branch-free activations: one v_exp_f32 + one v_rcp_f32 each, saturate at +-inf
__device__ __forceinline__ float fast_sigmoid(float x) {
    return __builtin_amdgcn_rcpf(1.0f + __expf(-x));
}
__device__ __forceinline__ float fast_tanh(float x) {
    float e = __expf(2.0f * x);
    return 1.0f - 2.0f * __builtin_amdgcn_rcpf(e + 1.0f);
}

// --- prep: x (f32) -> bf16, same [B][T][I] layout --------------------------
__global__ void convert_x_kernel(const float* __restrict__ x,
                                 unsigned short* __restrict__ xb, int n) {
    for (int i = blockIdx.x * blockDim.x + threadIdx.x; i < n;
         i += gridDim.x * blockDim.x)
        xb[i] = f32_to_bf16(x[i]);
}

// --- prep: W [GDIM][KDIM] f32 -> WMMA-B-ready bf16 packing -----------------
// Wb[((k/16)*GDIM + n)*16 + (k%16)] : each lane's 16-K fragment is one
// contiguous 32-byte chunk, coalesced across lanes.
__global__ void build_Wb_kernel(const float* __restrict__ W,
                                unsigned short* __restrict__ Wb) {
    const int total = GDIM * KDIM;
    for (int i = blockIdx.x * blockDim.x + threadIdx.x; i < total;
         i += gridDim.x * blockDim.x) {
        int n = i / KDIM;
        int k = i - n * KDIM;
        Wb[((size_t)(k >> 4) * GDIM + n) * 16 + (k & 15)] = f32_to_bf16(W[i]);
    }
}

// --- prep: zero h ping-pong buffers + grid-barrier counter ------------------
__global__ void init_state_kernel(unsigned short* __restrict__ hbuf,
                                  unsigned* __restrict__ counter) {
    const int n = 2 * BATCH * HDIM;
    for (int i = blockIdx.x * blockDim.x + threadIdx.x; i < n;
         i += gridDim.x * blockDim.x)
        hbuf[i] = 0;
    if (blockIdx.x == 0 && threadIdx.x == 0) *counter = 0u;
}

// --- persistent LSTM recurrence --------------------------------------------
__global__ __launch_bounds__(256) void lstm_persistent(
    const unsigned short* __restrict__ xb,    // [B][T][IDIM] bf16
    const unsigned short* __restrict__ Wb,    // packed [(KDIM/16)][GDIM][16] bf16
    const float*          __restrict__ bias,  // [GDIM] f32
    unsigned short*       __restrict__ hbuf,  // [2][B][HDIM] bf16 ping-pong
    unsigned*             __restrict__ counter,
    float*                __restrict__ out)   // [B][T][HDIM] f32
{
    __shared__ float gate_lds[4 * 32 * 16];   // [gate][batch][j] 8 KB
    __shared__ float c_lds[32 * 16];          // cell state slice 2 KB

    const int tid   = threadIdx.x;
    const int lane  = tid & 31;
    const int wave  = tid >> 5;        // 8 waves
    const int half  = lane >> 4;       // 0: lanes 0-15, 1: lanes 16-31
    const int l15   = lane & 15;
    const int mtile = wave & 1;        // batch tile: rows 0-15 / 16-31
    const int gidx  = wave >> 1;       // gate block 0..3 (i,f,g,o)
    const int j0    = blockIdx.x * 16; // this WG's 16 hidden units

    for (int p = tid; p < 32 * 16; p += 256) c_lds[p] = 0.0f;
    __syncthreads();

    const int b_row = mtile * 16 + l15;          // A-matrix row (batch) per lane
    const int n_col = gidx * HDIM + j0 + l15;    // B-matrix column per lane
    const float bias_v = bias[n_col];

    // Per-lane invariant bases.
    const unsigned short* xbase = xb + (size_t)b_row * SEQ * IDIM;     // +t*IDIM
    const unsigned short* wcol  = Wb + (size_t)n_col * 16;             // +blk*GDIM*16
    const size_t wblk = (size_t)GDIM * 16;                             // stride per 16-K block

    for (int t = 0; t < SEQ; ++t) {
        const int pr = t & 1;  // read buffer; write buffer is (t+1)&1
        const unsigned short* xrow = xbase + (size_t)t * IDIM;
        const unsigned short* hrow = hbuf + (size_t)pr * BATCH * HDIM
                                          + (size_t)b_row * HDIM;

        v8f acc;
        #pragma unroll
        for (int r = 0; r < 8; ++r) acc[r] = bias_v;

        // ---- x-part of K: K = 0..511 (16 WMMA) ----
        #pragma unroll 4
        for (int kk = 0; kk < IDIM / 32; ++kk) {
            const int kb = kk * 32;
            union { v16bf v; uint4 u[2]; } afrag, bfrag;
            // A (16x32 bf16): lane<16 holds K 0..15, lane>=16 holds K 16..31
            afrag.u[0] = *(const uint4*)(xrow + kb + half * 16);
            afrag.u[1] = *(const uint4*)(xrow + kb + half * 16 + 8);
            // B (32x16 bf16): lane<16 holds K 0..15, lane>=16 holds K 16..31
            const unsigned short* bsrc = wcol + (size_t)((kb >> 4) + half) * wblk;
            bfrag.u[0] = *(const uint4*)(bsrc);
            bfrag.u[1] = *(const uint4*)(bsrc + 8);
            acc = __builtin_amdgcn_wmma_f32_16x16x32_bf16(
                false, afrag.v, false, bfrag.v, (short)0, acc, false, false);
        }

        // ---- h-part of K: K = 512..1535 (32 WMMA) ----
        #pragma unroll 4
        for (int kk = 0; kk < HDIM / 32; ++kk) {
            const int kb = kk * 32;
            union { v16bf v; uint4 u[2]; } afrag, bfrag;
            afrag.u[0] = *(const uint4*)(hrow + kb + half * 16);
            afrag.u[1] = *(const uint4*)(hrow + kb + half * 16 + 8);
            const unsigned short* bsrc =
                wcol + (size_t)(((IDIM + kb) >> 4) + half) * wblk;
            bfrag.u[0] = *(const uint4*)(bsrc);
            bfrag.u[1] = *(const uint4*)(bsrc + 8);
            acc = __builtin_amdgcn_wmma_f32_16x16x32_bf16(
                false, afrag.v, false, bfrag.v, (short)0, acc, false, false);
        }

        // stage this wave's 16x16 gate tile: C layout -> (m = r + half*8, n = l15)
        #pragma unroll
        for (int r = 0; r < 8; ++r) {
            const int m = mtile * 16 + r + half * 8;
            gate_lds[(gidx * 32 + m) * 16 + l15] = acc[r];
        }
        __syncthreads();

        // elementwise LSTM cell update: 512 (b,j) pairs / 256 threads
        unsigned short* hnext = hbuf + (size_t)((t + 1) & 1) * BATCH * HDIM;
        #pragma unroll
        for (int p = tid; p < 32 * 16; p += 256) {
            const int bb = p >> 4, jj = p & 15;
            const float ig = gate_lds[(0 * 32 + bb) * 16 + jj];
            const float fg = gate_lds[(1 * 32 + bb) * 16 + jj];
            const float gg = gate_lds[(2 * 32 + bb) * 16 + jj];
            const float og = gate_lds[(3 * 32 + bb) * 16 + jj];
            const float cn = fast_sigmoid(fg) * c_lds[p]
                           + fast_sigmoid(ig) * fast_tanh(gg);
            const float hn = fast_sigmoid(og) * fast_tanh(cn);
            c_lds[p] = cn;
            out[((size_t)bb * SEQ + t) * HDIM + j0 + jj] = hn;
            hnext[(size_t)bb * HDIM + j0 + jj] = f32_to_bf16(hn);
        }
        __syncthreads();

        // grid-wide barrier: all WGs finish step t before step t+1 overwrites
        // the h buffer that was read this step.
        __threadfence();
        if (tid == 0) {
            atomicAdd(counter, 1u);
            const unsigned target = (unsigned)(t + 1) * NBLOCKS;
            while (__hip_atomic_load(counter, __ATOMIC_RELAXED,
                                     __HIP_MEMORY_SCOPE_AGENT) < target) {
                __builtin_amdgcn_s_sleep(2);
            }
        }
        __syncthreads();
        __threadfence();
    }
}

// ---------------------------------------------------------------------------
extern "C" void kernel_launch(void* const* d_in, const int* in_sizes, int n_in,
                              void* d_out, int out_size, void* d_ws, size_t ws_size,
                              hipStream_t stream) {
    const float* x = (const float*)d_in[0];   // [32][1024][512]
    const float* W = (const float*)d_in[1];   // [4096][1536]
    const float* b = (const float*)d_in[2];   // [4096]
    float* out = (float*)d_out;               // [32][1024][1024]

    // workspace layout (all 16B-aligned offsets)
    uint8_t* ws = (uint8_t*)d_ws;
    const size_t xb_bytes = (size_t)BATCH * SEQ * IDIM * 2;  // 32 MB
    const size_t wb_bytes = (size_t)KDIM * GDIM * 2;         // 12 MB
    const size_t hb_bytes = (size_t)2 * BATCH * HDIM * 2;    // 128 KB
    unsigned short* xb   = (unsigned short*)(ws);
    unsigned short* Wb   = (unsigned short*)(ws + xb_bytes);
    unsigned short* hbuf = (unsigned short*)(ws + xb_bytes + wb_bytes);
    unsigned* counter    = (unsigned*)(ws + xb_bytes + wb_bytes + hb_bytes);

    convert_x_kernel<<<2048, 256, 0, stream>>>(x, xb, BATCH * SEQ * IDIM);
    build_Wb_kernel<<<2048, 256, 0, stream>>>(W, Wb);
    init_state_kernel<<<256, 256, 0, stream>>>(hbuf, counter);
    lstm_persistent<<<NBLOCKS, 256, 0, stream>>>(xb, Wb, b, hbuf, counter, out);
}
